// TokenSparse_25374666785592
// MI455X (gfx1250) — matrix operand order repaired and checked by reading, not compiled, not run on hardware
//
#include <hip/hip_runtime.h>
#include <hip/hip_bf16.h>
#include <math.h>

#define BB 32
#define NN_ 4096
#define CC 512
#define HH 128
#define NKEEP 2048

typedef __attribute__((ext_vector_type(16))) __bf16 v16bf;
typedef __attribute__((ext_vector_type(8)))  float  v8f;
typedef __attribute__((ext_vector_type(16))) unsigned short u16x16;

__device__ __forceinline__ unsigned short f32_to_bf16_rne(float f) {
    unsigned int u = __builtin_bit_cast(unsigned int, f);
    unsigned int r = u + 0x7FFFu + ((u >> 16) & 1u);
    return (unsigned short)(r >> 16);
}

// ---------------------------------------------------------------------------
// Kernel 1: convert W1 (C x H fp32, row-major) into bf16 pre-swizzled into the
// WMMA B-fragment lane layout used here (contiguous k order, matching A):
//   g = (((kk*8 + j)*32) + lane)*16 + t
//   lane<16  holds W1[kk*32 +  0..15][j*16+lane]   (t = consecutive K)
//   lane>=16 holds W1[kk*32 + 16..31][j*16+lane-16]
// ---------------------------------------------------------------------------
__global__ void prep_w1_bf16(const float* __restrict__ W1,
                             unsigned short* __restrict__ out) {
    int g  = blockIdx.x * blockDim.x + threadIdx.x;   // 0..65535
    int t  =  g        & 15;
    int l  = (g >> 4)  & 31;
    int j  = (g >> 9)  & 7;
    int kk =  g >> 12;
    int k  = kk * 32 + ((l >= 16) ? 16 : 0) + t;
    int n  = j * 16 + (l & 15);
    out[g] = f32_to_bf16_rne(W1[k * HH + n]);
}

// ---------------------------------------------------------------------------
// Kernel 2: s_pred = sigmoid(gelu(tokens @ W1 + b1) @ W2 + b2)
// One wave per 16-row tile; 8 f32 accumulator tiles over H=128; K=512 in
// 16 steps of 32 via v_wmma_f32_16x16x32_bf16. W1 staged via LDS in 32KB
// quarters shared by all 8 waves of the block.
// ---------------------------------------------------------------------------
__global__ void __launch_bounds__(256)
score_pred_wmma(const float* __restrict__ tokens,
                const unsigned short* __restrict__ w1bf,
                const float* __restrict__ b1,
                const float* __restrict__ W2,
                const float* __restrict__ b2,
                float* __restrict__ s_pred) {
    __shared__ unsigned short w1s[16384];   // 32 KB = one quarter of W1(bf16)

    const int lane   = threadIdx.x & 31;
    const int wave   = threadIdx.x >> 5;
    const int ln     = lane & 15;
    const int hiHalf = lane >> 4;           // 0 or 1

    const int NT = (BB * NN_) / 16;         // 8192 row tiles
    for (int tbase = blockIdx.x * 8; tbase < NT; tbase += gridDim.x * 8) {
        const int tile    = tbase + wave;   // grid sized so tile < NT
        const int rowBase = tile * 16;
        const float* Arow = tokens + (size_t)(rowBase + ln) * CC;

        v8f acc[8];
        const v8f vzero = {};
        #pragma unroll
        for (int j = 0; j < 8; ++j) acc[j] = vzero;

        for (int q = 0; q < 4; ++q) {       // quarter of K (128 of 512)
            __syncthreads();
            const uint4* src = (const uint4*)w1bf + q * 2048;
            uint4* dst = (uint4*)w1s;
            for (int i = threadIdx.x; i < 2048; i += 256) dst[i] = src[i];
            __syncthreads();

            #pragma unroll
            for (int kq = 0; kq < 4; ++kq) {
                const int kk = q * 4 + kq;
                const int kb = kk * 32 + hiHalf * 16;
                // A fragment: lane<16 -> K {0..15}, lane>=16 -> {16..31}
                float4 f0 = *(const float4*)(Arow + kb + 0);
                float4 f1 = *(const float4*)(Arow + kb + 4);
                float4 f2 = *(const float4*)(Arow + kb + 8);
                float4 f3 = *(const float4*)(Arow + kb + 12);
                u16x16 au;
                au[0]  = f32_to_bf16_rne(f0.x);  au[1]  = f32_to_bf16_rne(f0.y);
                au[2]  = f32_to_bf16_rne(f0.z);  au[3]  = f32_to_bf16_rne(f0.w);
                au[4]  = f32_to_bf16_rne(f1.x);  au[5]  = f32_to_bf16_rne(f1.y);
                au[6]  = f32_to_bf16_rne(f1.z);  au[7]  = f32_to_bf16_rne(f1.w);
                au[8]  = f32_to_bf16_rne(f2.x);  au[9]  = f32_to_bf16_rne(f2.y);
                au[10] = f32_to_bf16_rne(f2.z);  au[11] = f32_to_bf16_rne(f2.w);
                au[12] = f32_to_bf16_rne(f3.x);  au[13] = f32_to_bf16_rne(f3.y);
                au[14] = f32_to_bf16_rne(f3.z);  au[15] = f32_to_bf16_rne(f3.w);
                v16bf a = __builtin_bit_cast(v16bf, au);

                #pragma unroll
                for (int j = 0; j < 8; ++j) {
                    const u16x16 bu =
                        *(const u16x16*)&w1s[(((kq * 8 + j) * 32) + lane) * 16];
                    v16bf bfrag = __builtin_bit_cast(v16bf, bu);
                    acc[j] = __builtin_amdgcn_wmma_f32_16x16x32_bf16(
                        false, a, false, bfrag, (short)0, acc[j], false, false);
                }
            }
        }

        // Epilogue: gelu(h + b1) dot W2, reduced across 16-lane halves.
        float p[8];
        #pragma unroll
        for (int r = 0; r < 8; ++r) p[r] = 0.0f;
        #pragma unroll
        for (int j = 0; j < 8; ++j) {
            const int hcol = j * 16 + ln;
            const float b1v = b1[hcol];
            const float w2v = W2[hcol];
            #pragma unroll
            for (int r = 0; r < 8; ++r) {
                float x = acc[j][r] + b1v;
                float g = 0.5f * x * (1.0f + erff(x * 0.7071067811865475f));
                p[r] += g * w2v;
            }
        }
        for (int off = 1; off < 16; off <<= 1) {
            #pragma unroll
            for (int r = 0; r < 8; ++r) p[r] += __shfl_xor(p[r], off, 32);
        }
        if (ln == 0) {
            const float b2v = b2[0];
            const int rb = rowBase + hiHalf * 8;   // C/D layout: halves = rows 0-7 / 8-15
            #pragma unroll
            for (int r = 0; r < 8; ++r) {
                float z = p[r] + b2v;
                s_pred[rb + r] = 1.0f / (1.0f + expf(-z));
            }
        }
    }
}

// ---------------------------------------------------------------------------
// Kernel 3: per-batch min/max normalization + score fusion.
// ---------------------------------------------------------------------------
__global__ void __launch_bounds__(256)
combine_scores(const float* __restrict__ sa, const float* __restrict__ m2,
               const float* __restrict__ m3, const float* __restrict__ s_pred,
               float* __restrict__ score) {
    __shared__ float red[6][256];
    const int b = blockIdx.x, tid = threadIdx.x;
    const float* A = sa + b * NN_;
    const float* X = m2 + b * NN_;
    const float* Y = m3 + b * NN_;
    float mnA = INFINITY, mxA = -INFINITY;
    float mnX = INFINITY, mxX = -INFINITY;
    float mnY = INFINITY, mxY = -INFINITY;
    for (int n = tid; n < NN_; n += 256) {
        float a = A[n], x = X[n], y = Y[n];
        mnA = fminf(mnA, a); mxA = fmaxf(mxA, a);
        mnX = fminf(mnX, x); mxX = fmaxf(mxX, x);
        mnY = fminf(mnY, y); mxY = fmaxf(mxY, y);
    }
    red[0][tid] = mnA; red[1][tid] = mxA;
    red[2][tid] = mnX; red[3][tid] = mxX;
    red[4][tid] = mnY; red[5][tid] = mxY;
    __syncthreads();
    for (int s = 128; s > 0; s >>= 1) {
        if (tid < s) {
            red[0][tid] = fminf(red[0][tid], red[0][tid + s]);
            red[1][tid] = fmaxf(red[1][tid], red[1][tid + s]);
            red[2][tid] = fminf(red[2][tid], red[2][tid + s]);
            red[3][tid] = fmaxf(red[3][tid], red[3][tid + s]);
            red[4][tid] = fminf(red[4][tid], red[4][tid + s]);
            red[5][tid] = fmaxf(red[5][tid], red[5][tid + s]);
        }
        __syncthreads();
    }
    const float oA = red[0][0], sA = 1.0f / (red[1][0] - red[0][0] + 1e-8f);
    const float oX = red[2][0], sX = 1.0f / (red[3][0] - red[2][0] + 1e-8f);
    const float oY = red[4][0], sY = 1.0f / (red[5][0] - red[4][0] + 1e-8f);
    for (int n = tid; n < NN_; n += 256) {
        float nim = (A[n] - oA) * sA;
        float nm2 = (X[n] - oX) * sX;
        float nm3 = (Y[n] - oY) * sY;
        score[b * NN_ + n] =
            0.5f * s_pred[b * NN_ + n] + 0.25f * (nm2 + nm3 + 2.0f * nim);
    }
}

// ---------------------------------------------------------------------------
// Kernel 4: per-batch bitonic argsort (descending, index tie-break) in LDS,
// then emit keep_policy / selected_mask / score_mask + sorted arrays.
// ---------------------------------------------------------------------------
__global__ void __launch_bounds__(1024)
sort_and_masks(const float* __restrict__ score,
               int* __restrict__ sortedIdx, float* __restrict__ sortedScore,
               float* __restrict__ out_keep, float* __restrict__ out_selmask,
               float* __restrict__ out_mask) {
    __shared__ float sk[NN_];
    __shared__ int   si[NN_];
    const int b = blockIdx.x, tid = threadIdx.x;
    for (int n = tid; n < NN_; n += 1024) { sk[n] = score[b * NN_ + n]; si[n] = n; }
    __syncthreads();
    for (int k = 2; k <= NN_; k <<= 1) {
        for (int j = k >> 1; j > 0; j >>= 1) {
            for (int t = tid; t < NN_ / 2; t += 1024) {
                int i = ((t & ~(j - 1)) << 1) | (t & (j - 1));
                int q = i + j;
                bool desc = ((i & k) == 0);
                float ki = sk[i], kq = sk[q];
                int   ii = si[i], iq = si[q];
                // "i before q" order: higher score first; ties -> lower index first
                bool iFirst = (ki > kq) || (ki == kq && ii < iq);
                bool doSwap = desc ? (!iFirst) : iFirst;
                if (doSwap) { sk[i] = kq; sk[q] = ki; si[i] = iq; si[q] = ii; }
            }
            __syncthreads();
        }
    }
    for (int p = tid; p < NN_; p += 1024) {
        float s = sk[p]; int ix = si[p];
        sortedScore[b * NN_ + p] = s;
        sortedIdx[b * NN_ + p]   = ix;
        out_mask[(size_t)b * NN_ + ix] = (p < NKEEP) ? 1.0f : 0.0f;
        if (p < NKEEP) {
            out_keep[b * NKEEP + p]    = (float)ix;
            out_selmask[b * NKEEP + p] = 1.0f;
        }
    }
}

// ---------------------------------------------------------------------------
// Kernel 5: gather kept token rows (coalesced float4).
// ---------------------------------------------------------------------------
__global__ void __launch_bounds__(128)
gather_tokens(const float* __restrict__ tokens, const int* __restrict__ sortedIdx,
              float* __restrict__ out_sel) {
    int bp = blockIdx.x;
    int b  = bp >> 11;             // / NKEEP
    int p  = bp & (NKEEP - 1);
    int src = sortedIdx[b * NN_ + p];
    const float4* s = (const float4*)(tokens + ((size_t)b * NN_ + src) * CC);
    float4* d = (float4*)(out_sel + ((size_t)b * NKEEP + p) * CC);
    d[threadIdx.x] = s[threadIdx.x];
}

// ---------------------------------------------------------------------------
// Kernel 6: softmax over non-kept (sorted tail) + weighted token sum.
// ---------------------------------------------------------------------------
__global__ void __launch_bounds__(256)
extra_token_kernel(const float* __restrict__ tokens,
                   const int* __restrict__ sortedIdx,
                   const float* __restrict__ sortedScore,
                   float* __restrict__ out_extra) {
    __shared__ float wlds[NN_ - NKEEP];
    __shared__ int   ilds[NN_ - NKEEP];
    __shared__ float zred[256];
    const int b = blockIdx.x, tid = threadIdx.x;
    const int NT = NN_ - NKEEP;                       // 2048
    const float m = sortedScore[b * NN_ + NKEEP];     // tail max (desc sorted)
    float zs = 0.0f;
    for (int p = tid; p < NT; p += 256) {
        float e = expf(sortedScore[b * NN_ + NKEEP + p] - m);
        wlds[p] = e;
        ilds[p] = sortedIdx[b * NN_ + NKEEP + p];
        zs += e;
    }
    zred[tid] = zs; __syncthreads();
    for (int s = 128; s > 0; s >>= 1) {
        if (tid < s) zred[tid] += zred[tid + s];
        __syncthreads();
    }
    const float invZ = 1.0f / zred[0];
    float acc0 = 0.0f, acc1 = 0.0f;
    const int c0 = tid, c1 = tid + 256;
    const float* tb = tokens + (size_t)b * NN_ * CC;
    for (int p = 0; p < NT; ++p) {
        const float w = wlds[p];
        const float* row = tb + (size_t)ilds[p] * CC;
        acc0 += w * row[c0];
        acc1 += w * row[c1];
    }
    out_extra[b * CC + c0] = acc0 * invZ;
    out_extra[b * CC + c1] = acc1 * invZ;
}

// ---------------------------------------------------------------------------
extern "C" void kernel_launch(void* const* d_in, const int* in_sizes, int n_in,
                              void* d_out, int out_size, void* d_ws, size_t ws_size,
                              hipStream_t stream) {
    (void)in_sizes; (void)n_in; (void)out_size; (void)ws_size;
    const float* tokens   = (const float*)d_in[0];
    const float* self_att = (const float*)d_in[1];
    const float* m2       = (const float*)d_in[2];
    const float* m3       = (const float*)d_in[3];
    const float* W1       = (const float*)d_in[4];
    const float* b1       = (const float*)d_in[5];
    const float* W2       = (const float*)d_in[6];
    const float* b2       = (const float*)d_in[7];

    char* ws = (char*)d_ws;
    unsigned short* w1bf = (unsigned short*)ws;                       // 128 KB
    float* s_pred        = (float*)(ws + 131072);                     // 512 KB
    float* score         = (float*)(ws + 131072 + 1 * 524288);        // 512 KB
    float* sortedScore   = (float*)(ws + 131072 + 2 * 524288);        // 512 KB
    int*   sortedIdx     = (int*)  (ws + 131072 + 3 * 524288);        // 512 KB

    float* out       = (float*)d_out;
    float* out_sel   = out;                                           // B*NK*C
    float* out_extra = out + (size_t)BB * NKEEP * CC;                 // B*C
    float* out_mask  = out_extra + BB * CC;                           // B*N
    float* out_selm  = out_mask + BB * NN_;                           // B*NK
    float* out_keep  = out_selm + BB * NKEEP;                         // B*NK

    prep_w1_bf16      <<<256, 256, 0, stream>>>(W1, w1bf);
    score_pred_wmma   <<<1024, 256, 0, stream>>>(tokens, w1bf, b1, W2, b2, s_pred);
    combine_scores    <<<BB, 256, 0, stream>>>(self_att, m2, m3, s_pred, score);
    sort_and_masks    <<<BB, 1024, 0, stream>>>(score, sortedIdx, sortedScore,
                                                out_keep, out_selm, out_mask);
    gather_tokens     <<<BB * NKEEP, 128, 0, stream>>>(tokens, sortedIdx, out_sel);
    extra_token_kernel<<<BB, 256, 0, stream>>>(tokens, sortedIdx, sortedScore,
                                               out_extra);
}
